// SiLUMixAndBias_34772055229039
// MI455X (gfx1250) — compile-verified, matrix operand order checked by reading, not run on hardware
//
#include <hip/hip_runtime.h>

typedef __attribute__((ext_vector_type(16))) _Float16 v16h;
typedef __attribute__((ext_vector_type(8)))  _Float16 v8h;
typedef __attribute__((ext_vector_type(4)))  _Float16 v4h;
typedef __attribute__((ext_vector_type(8)))  float    v8f;
typedef __attribute__((ext_vector_type(4)))  float    v4f;

#define KK 16          // spectral components
#define DD 256         // feature dim
#define TM 16          // nodes (rows) per workgroup / WMMA M tile
#define ROWH (DD + 8)  // padded f16 row stride: 264 halfs = 528 B -> 4-bank rotate/row

__global__ __launch_bounds__(256) void silu_gate_kernel(
    const float* __restrict__ x, const float* __restrict__ W,
    const float* __restrict__ bias, float* __restrict__ out)
{
    __shared__ _Float16 norm_h[TM * ROWH];   // WMMA A-matrix (f16), padded rows
    __shared__ float    gate[TM][DD];        // sigmoid gate (f32)

    const int t    = threadIdx.x;            // 0..255
    const int n0   = blockIdx.x * TM;        // first node of this tile
    const int wave = t >> 5;                 // 0..7
    const int lane = t & 31;
    const int m    = lane & 15;              // row within 16x16 fragment
    const int hi   = lane >> 4;              // K-half selector (A/B layout)

    // b128 tiling for the streaming phases: thread owns 4 consecutive columns
    // (d or e) for 4 consecutive nodes.
    const int tq = t & 63;                   // column-quad id
    const int g  = t >> 6;                   // node-group id (0..3)
    const int c0 = tq * 4;                   // first column (d0 / e0)

    // Prefetch the W rows this wave's WMMAs will consume (global_prefetch_b8);
    // they stream into L2/WGP$ while phase 1 chews through x.
    __builtin_prefetch(W + (size_t)(wave * 16 + m) * DD, 0, 0);
    __builtin_prefetch(W + (size_t)((wave + 8) * 16 + m) * DD, 0, 0);

    // ---- Phase 1: norm[n,d] = sqrt(sum_k x[n,k,d]^2). float4 (b128) loads,
    //      f32 accumulate in regs, f16 b64 stores into the LDS A-matrix.
    {
        v4f acc[4];
#pragma unroll
        for (int ni = 0; ni < 4; ++ni) acc[ni] = (v4f){0.f, 0.f, 0.f, 0.f};
        const float* xp = x + (size_t)n0 * KK * DD + c0;
#pragma unroll
        for (int ni = 0; ni < 4; ++ni) {
            const int n = g * 4 + ni;
            const float* xr = xp + (size_t)n * KK * DD;
#pragma unroll
            for (int k = 0; k < KK; ++k) {
                const v4f v = *(const v4f*)(xr + (size_t)k * DD);
                acc[ni] += v * v;
            }
        }
#pragma unroll
        for (int ni = 0; ni < 4; ++ni) {
            const int n = g * 4 + ni;
            v4h h;
#pragma unroll
            for (int c = 0; c < 4; ++c) h[c] = (_Float16)sqrtf(acc[ni][c]);
            *(v4h*)&norm_h[n * ROWH + c0] = h;
        }
    }
    __syncthreads();

    // ---- Phase 2: gate_pre[16,256] = norm(16x256) @ W^T via v_wmma_f32_16x16x32_f16.
    //      A fragments are shared by both e-tiles -> load them from LDS once.
    v16h afrag[DD / 32];
#pragma unroll
    for (int dc = 0; dc < DD / 32; ++dc) {
        const int kb = dc * 32 + hi * 8;
        // ISA 16-bit A 16x32 layout:
        //  lanes 0-15:  elems 0-7 = K 0-7,  elems 8-15 = K 16-23
        //  lanes 16-31: elems 0-7 = K 8-15, elems 8-15 = K 24-31
        const v8h alo = *(const v8h*)&norm_h[m * ROWH + kb];
        const v8h ahi = *(const v8h*)&norm_h[m * ROWH + kb + 16];
#pragma unroll
        for (int i = 0; i < 8; ++i) { afrag[dc][i] = alo[i]; afrag[dc][i + 8] = ahi[i]; }
    }

#pragma unroll
    for (int tt = 0; tt < 2; ++tt) {
        const int et = wave + tt * 8;          // e-tile index 0..15
        const int e  = et * 16 + m;            // this lane's output column
        const float* wrow = W + (size_t)e * DD;

        v8f c = {};
#pragma unroll
        for (int dc = 0; dc < DD / 32; ++dc) {
            const int kb = dc * 32 + hi * 8;
            // B fragment: B[k][n] = W[e=n][d=k] -> contiguous along d in each
            // lane's own W row; f32 -> f16 converted in registers.
            const v4f w0 = *(const v4f*)(wrow + kb);
            const v4f w1 = *(const v4f*)(wrow + kb + 4);
            const v4f w2 = *(const v4f*)(wrow + kb + 16);
            const v4f w3 = *(const v4f*)(wrow + kb + 20);
            v16h bf;
#pragma unroll
            for (int i = 0; i < 4; ++i) {
                bf[i]      = (_Float16)w0[i];
                bf[i + 4]  = (_Float16)w1[i];
                bf[i + 8]  = (_Float16)w2[i];
                bf[i + 12] = (_Float16)w3[i];
            }
            c = __builtin_amdgcn_wmma_f32_16x16x32_f16(
                    /*neg_a=*/false, afrag[dc], /*neg_b=*/false, bf,
                    /*c_mod=*/(short)0, c, /*reuse_a=*/false, /*reuse_b=*/false);
        }

        // ---- Phase 3: bias + sigmoid on the f32 D fragment, scatter to LDS.
        //      D layout: VGPR j, lanes 0-15 -> M=j, lanes 16-31 -> M=j+8; N=lane&15.
        const float bv = bias[e];
#pragma unroll
        for (int j = 0; j < 8; ++j) {
            const int row = j + hi * 8;
            const float gv = c[j] + bv;
            gate[row][e] = 1.0f / (1.0f + __expf(-gv));
        }
    }
    __syncthreads();

    // ---- Phase 4: out[n,k,e] = x[n,k,e] * gate[n,e]. b128 loads of the
    //      L2-warm x tile, b128 non-temporal stores of out.
    {
#pragma unroll
        for (int ni = 0; ni < 4; ++ni) {
            const int n = g * 4 + ni;
            const v4f gv = *(const v4f*)&gate[n][c0];
            const size_t base = (size_t)(n0 + n) * KK * DD + c0;
#pragma unroll
            for (int k = 0; k < KK; ++k) {
                const v4f xv = *(const v4f*)(x + base + (size_t)k * DD);
                const v4f ov = xv * gv;
                __builtin_nontemporal_store(ov, (v4f*)(out + base + (size_t)k * DD));
            }
        }
    }
}

extern "C" void kernel_launch(void* const* d_in, const int* in_sizes, int n_in,
                              void* d_out, int out_size, void* d_ws, size_t ws_size,
                              hipStream_t stream) {
    const float* x = (const float*)d_in[0];   // [N, K, D] f32
    const float* W = (const float*)d_in[1];   // [D, D]   f32
    const float* b = (const float*)d_in[2];   // [D]      f32
    float* out = (float*)d_out;               // [N, K, D] f32

    const int N = in_sizes[0] / (KK * DD);    // 50000 (divisible by TM=16)
    const int nblocks = (N + TM - 1) / TM;    // 3125
    silu_gate_kernel<<<dim3(nblocks), dim3(256), 0, stream>>>(x, W, b, out);
}
